// AdjacencyGenerator_61074434949406
// MI455X (gfx1250) — matrix operand order repaired and checked by reading, not compiled, not run on hardware
//
#include <hip/hip_runtime.h>

typedef __bf16 bf16;
typedef __attribute__((ext_vector_type(16))) __bf16 v16bf;
typedef __attribute__((ext_vector_type(8)))  __bf16 v8bf;
typedef __attribute__((ext_vector_type(8)))  float  v8f;

#define DM 256
#define HID 768
#define NE 160000
#define NN 10000
#define EPS 1e-5f

// ---------------- elementwise f32 -> bf16 ----------------
__global__ __launch_bounds__(256)
void k_cvt(const float* __restrict__ s, bf16* __restrict__ d, int n) {
  int i = blockIdx.x * 256 + threadIdx.x;
  if (i < n) d[i] = (bf16)s[i];
}

// ---------------- gather x[src] -> bf16, x[dst] -> f32+bf16 ----------------
__global__ __launch_bounds__(256)
void k_gather(const int* __restrict__ src, const int* __restrict__ dst,
              const float* __restrict__ x,
              bf16* __restrict__ xib, float* __restrict__ qf, bf16* __restrict__ qb) {
  const int e = blockIdx.x, d = threadIdx.x;
  const size_t o = (size_t)e * DM + d;
  const float xs = x[(size_t)src[e] * DM + d];
  const float xd = x[(size_t)dst[e] * DM + d];
  xib[o] = (bf16)xs;
  qf[o]  = xd;
  qb[o]  = (bf16)xd;
}

// ---------------- WMMA GEMM: C[M,N] = A[M,K](bf16) @ W[N,K]^T(bf16) + bias ----
// block = 256 threads = 8 waves; wave -> 16(M) x 128(N) tile (8 acc tiles);
// grid = (N/128, M/128). K is a template constant (256 or 768) so that after
// full unroll every A/B fragment load is base-pointer + compile-time immediate
// offset: zero address arithmetic in the hot loop, VGPRs free for in-flight
// fragments (load/WMMA double buffering).
template<int K>
__global__ __launch_bounds__(256)
void k_gemm(const bf16* __restrict__ A, const bf16* __restrict__ W,
            const float* __restrict__ bias,
            float* __restrict__ Cf, bf16* __restrict__ Cb,
            int N, int relu) {
  const int lane = threadIdx.x & 31;
  const int wave = threadIdx.x >> 5;
  const int m0 = blockIdx.y * 128 + wave * 16;
  const int n0 = blockIdx.x * 128;
  const int aoff = (lane >> 4) * 8;    // A frag: half-lanes split K by 8 within each 16
  const int boff = (lane >> 4) * 8;    // B frag: half-lanes split K by 8 within each 16
  const bf16* ap = A + (size_t)(m0 + (lane & 15)) * K + aoff;
  const bf16* wp = W + (size_t)(n0 + (lane & 15)) * K + boff;

  v8f acc[8];
  #pragma unroll
  for (int n = 0; n < 8; ++n) acc[n] = (v8f){};

  #pragma unroll
  for (int k0 = 0; k0 < K; k0 += 32) {
    v8bf alo = *(const v8bf*)(ap + k0);
    v8bf ahi = *(const v8bf*)(ap + k0 + 16);
    v16bf av = __builtin_shufflevector(alo, ahi,
                 0,1,2,3,4,5,6,7,8,9,10,11,12,13,14,15);
    #pragma unroll
    for (int n = 0; n < 8; ++n) {
      v8bf bl = *(const v8bf*)(wp + n * 16 * K + k0);      // imm offset
      v8bf bh = *(const v8bf*)(wp + n * 16 * K + k0 + 16); // imm offset
      v16bf bv = __builtin_shufflevector(bl, bh,
                   0,1,2,3,4,5,6,7,8,9,10,11,12,13,14,15);
      acc[n] = __builtin_amdgcn_wmma_f32_16x16x32_bf16(
                   false, av, false, bv, (short)0, acc[n], false, false);
    }
  }

  const int rbase = (lane >> 4) * 8;   // C/D: VGPR r -> row r (lanes<16) / r+8
  const int c = lane & 15;
  #pragma unroll
  for (int n = 0; n < 8; ++n) {
    const int col = n0 + n * 16 + c;
    const float bv = bias[col];
    #pragma unroll
    for (int r = 0; r < 8; ++r) {
      float val = acc[n][r] + bv;
      if (relu) val = val > 0.f ? val : 0.2f * val;
      const size_t idx = (size_t)(m0 + rbase + r) * N + col;
      if (Cf) Cf[idx] = val;
      if (Cb) Cb[idx] = (bf16)val;
    }
  }
}

// ---------------- segment softmax helpers ----------------
__global__ __launch_bounds__(256)
void k_seginit(unsigned* __restrict__ amaxk, float* __restrict__ denom, int n) {
  int i = blockIdx.x * 256 + threadIdx.x;
  if (i < n) { amaxk[i] = 0u; denom[i] = 0.f; }
}

// one wave per edge: alpha = dot(q,k); atomicMax on order-preserving uint key
__global__ __launch_bounds__(256)
void k_alpha(const float* __restrict__ q, const float* __restrict__ k,
             const int* __restrict__ dst, float* __restrict__ alpha,
             unsigned* __restrict__ amaxk) {
  const int lane = threadIdx.x & 31;
  const int e = blockIdx.x * 8 + (threadIdx.x >> 5);
  const float4* qp = (const float4*)(q + (size_t)e * DM);
  const float4* kp = (const float4*)(k + (size_t)e * DM);
  float s = 0.f;
  #pragma unroll
  for (int i = 0; i < DM / 4 / 32; ++i) {
    float4 a = qp[lane + i * 32], b = kp[lane + i * 32];
    s += a.x * b.x + a.y * b.y + a.z * b.z + a.w * b.w;
  }
  #pragma unroll
  for (int off = 16; off > 0; off >>= 1) s += __shfl_xor(s, off, 32);
  if (lane == 0) {
    alpha[e] = s;
    unsigned b = __float_as_uint(s);
    unsigned key = (b & 0x80000000u) ? ~b : (b | 0x80000000u);
    atomicMax(amaxk + dst[e], key);
  }
}

__global__ __launch_bounds__(256)
void k_expsum(float* __restrict__ ev, const int* __restrict__ dst,
              const unsigned* __restrict__ amaxk, float* __restrict__ denom, int n) {
  int e = blockIdx.x * 256 + threadIdx.x;
  if (e >= n) return;
  unsigned key = amaxk[dst[e]];
  unsigned b = (key & 0x80000000u) ? (key & 0x7fffffffu) : ~key;
  float v = expf(ev[e] - __uint_as_float(b));
  ev[e] = v;
  atomicAdd(denom + dst[e], v);
}

// ---------------- block (256-thread) reductions / layernorm ----------------
__device__ __forceinline__ float blockSum256(float v, float* sh) {
  #pragma unroll
  for (int off = 16; off > 0; off >>= 1) v += __shfl_xor(v, off, 32);
  if ((threadIdx.x & 31) == 0) sh[threadIdx.x >> 5] = v;
  __syncthreads();
  float t = 0.f;
  #pragma unroll
  for (int i = 0; i < 8; ++i) t += sh[i];
  __syncthreads();
  return t;
}

__device__ __forceinline__ float ln256(float t, const float* g, const float* b,
                                       int d, float* sh) {
  const float m = blockSum256(t, sh) * (1.f / DM);
  const float c = t - m;
  const float var = blockSum256(c * c, sh) * (1.f / DM);
  return c * rsqrtf(var + EPS) * g[d] + b[d];
}

// attn_out = q + (e/denom[dst]) * v, then LN -> bf16 (next GEMM input)
__global__ __launch_bounds__(256)
void k_attn_ln(const float* __restrict__ q, const float* __restrict__ v,
               const float* __restrict__ ev, const float* __restrict__ denom,
               const int* __restrict__ dst, const float* __restrict__ g,
               const float* __restrict__ b, bf16* __restrict__ outb) {
  __shared__ float sh[8];
  const int e = blockIdx.x, d = threadIdx.x;
  const size_t o = (size_t)e * DM + d;
  const float attn = ev[e] / denom[dst[e]];
  const float t = q[o] + attn * v[o];
  outb[o] = (bf16)ln256(t, g, b, d, sh);
}

// lrelu + LN(gf,bf): overwrite f32 query, also emit bf16 for W3 GEMM
__global__ __launch_bounds__(256)
void k_post_ln(float* __restrict__ qf, const float* __restrict__ g,
               const float* __restrict__ b, bf16* __restrict__ outb) {
  __shared__ float sh[8];
  const int e = blockIdx.x, d = threadIdx.x;
  const size_t o = (size_t)e * DM + d;
  float t = qf[o];
  t = t > 0.f ? t : 0.2f * t;
  const float y = ln256(t, g, b, d, sh);
  qf[o] = y;
  outb[o] = (bf16)y;
}

// h = LN(h3 + query); out[e] = dot(h, Wvec) + bvec
__global__ __launch_bounds__(256)
void k_final(const float* __restrict__ h3, const float* __restrict__ qf,
             const float* __restrict__ g, const float* __restrict__ b,
             const float* __restrict__ wvec, const float* __restrict__ bvec,
             float* __restrict__ out) {
  __shared__ float sh[8];
  const int e = blockIdx.x, d = threadIdx.x;
  const size_t o = (size_t)e * DM + d;
  const float y = ln256(h3[o] + qf[o], g, b, d, sh);
  const float tot = blockSum256(y * wvec[d], sh);
  if (d == 0) out[e] = tot + bvec[0];
}

extern "C" void kernel_launch(void* const* d_in, const int* in_sizes, int n_in,
                              void* d_out, int out_size, void* d_ws, size_t ws_size,
                              hipStream_t stream) {
  (void)in_sizes; (void)n_in; (void)out_size; (void)ws_size;
  const int*   edge = (const int*)d_in[0];
  const int*   src  = edge;
  const int*   dst  = edge + NE;
  const float* x    = (const float*)d_in[1];
  const float* Wq   = (const float*)d_in[2];  const float* bq  = (const float*)d_in[3];
  const float* Wk   = (const float*)d_in[4];  const float* bk  = (const float*)d_in[5];
  const float* Wv   = (const float*)d_in[6];  const float* bv  = (const float*)d_in[7];
  const float* Wff  = (const float*)d_in[8];  const float* bff = (const float*)d_in[9];
  const float* ga   = (const float*)d_in[10]; const float* ba  = (const float*)d_in[11];
  const float* gfp  = (const float*)d_in[12]; const float* bfp = (const float*)d_in[13];
  const float* gfin = (const float*)d_in[14]; const float* bfin= (const float*)d_in[15];
  const float* W3   = (const float*)d_in[16]; const float* b3  = (const float*)d_in[17];
  const float* W4   = (const float*)d_in[18]; const float* b4  = (const float*)d_in[19];
  const float* W5   = (const float*)d_in[20]; const float* b5  = (const float*)d_in[21];
  const float* Wvec = (const float*)d_in[22]; const float* bvc = (const float*)d_in[23];
  float* out = (float*)d_out;

  const size_t E = NE, D = DM, H = HID;
  char* p = (char*)d_ws;
  auto carve = [&](size_t bytes) { char* r = p; p += (bytes + 255) & ~(size_t)255; return r; };
  float*    qf    = (float*)carve(E * D * 4);   // persistent query (f32)
  bf16*     xib   = (bf16*) carve(E * D * 2);   // x[src] bf16 (both layers)
  bf16*     B1    = (bf16*) carve(E * D * 2);   // bf16 ping
  bf16*     B2    = (bf16*) carve(E * D * 2);   // bf16 pong
  char*     DEF   =         carve(E * H * 4);   // q/k/v (layers) -> h2bf/h3 (head)
  float*    qbuf  = (float*)DEF;
  float*    kbuf  = qbuf + E * D;
  float*    vbuf  = qbuf + 2 * E * D;
  bf16*     h1bf  = (bf16*) carve(E * H * 2);
  float*    evb   = (float*)carve(E * 4);       // alpha, then exp values (in place)
  unsigned* amaxk = (unsigned*)carve(NN * 4);
  float*    denom = (float*)carve(NN * 4);
  bf16* WqB  = (bf16*)carve(2 * D * D * 2);
  bf16* WkB  = (bf16*)carve(2 * D * D * 2);
  bf16* WvB  = (bf16*)carve(2 * D * D * 2);
  bf16* WffB = (bf16*)carve(2 * D * D * 2);
  bf16* W3B  = (bf16*)carve(H * D * 2);
  bf16* W4B  = (bf16*)carve(H * H * 2);
  bf16* W5B  = (bf16*)carve(D * H * 2);
  bf16*  h2bf = (bf16*)DEF;                     // reuse q/k/v space (dead in head)
  float* h3   = (float*)(DEF + E * H * 2);      // tail of DEF region

  auto cvt = [&](const float* s, bf16* d, int n) {
    k_cvt<<<(n + 255) / 256, 256, 0, stream>>>(s, d, n);
  };
  cvt(Wq,  WqB,  2 * DM * DM);
  cvt(Wk,  WkB,  2 * DM * DM);
  cvt(Wv,  WvB,  2 * DM * DM);
  cvt(Wff, WffB, 2 * DM * DM);
  cvt(W3,  W3B,  HID * DM);
  cvt(W4,  W4B,  HID * HID);
  cvt(W5,  W5B,  DM * HID);

  k_gather<<<NE, 256, 0, stream>>>(src, dst, x, xib, qf, B1);

  const dim3 gD(DM / 128, NE / 128);   // (2, 1250)
  const dim3 gH(HID / 128, NE / 128);  // (6, 1250)

  const bf16* qin = B1;
  for (int l = 0; l < 2; ++l) {
    const size_t wo = (size_t)l * D * D;
    k_gemm<256><<<gD, 256, 0, stream>>>(qin, WqB + wo, bq + l * D, qbuf, nullptr, DM, 0);
    k_gemm<256><<<gD, 256, 0, stream>>>(xib, WkB + wo, bk + l * D, kbuf, nullptr, DM, 0);
    k_gemm<256><<<gD, 256, 0, stream>>>(xib, WvB + wo, bv + l * D, vbuf, nullptr, DM, 0);
    k_seginit<<<(NN + 255) / 256, 256, 0, stream>>>(amaxk, denom, NN);
    k_alpha<<<NE / 8, 256, 0, stream>>>(qbuf, kbuf, dst, evb, amaxk);
    k_expsum<<<(NE + 255) / 256, 256, 0, stream>>>(evb, dst, amaxk, denom, NE);
    k_attn_ln<<<NE, 256, 0, stream>>>(qbuf, vbuf, evb, denom, dst,
                                      ga + l * D, ba + l * D, B1);
    k_gemm<256><<<gD, 256, 0, stream>>>(B1, WffB + wo, bff + l * D, qf,
                                        (l == 0) ? B2 : nullptr, DM, 0);
    qin = B2;
  }

  k_post_ln<<<NE, 256, 0, stream>>>(qf, gfp, bfp, B1);
  k_gemm<256><<<gH, 256, 0, stream>>>(B1,   W3B, b3, nullptr, h1bf, HID, 1);
  k_gemm<768><<<gH, 256, 0, stream>>>(h1bf, W4B, b4, nullptr, h2bf, HID, 1);
  k_gemm<768><<<gD, 256, 0, stream>>>(h2bf, W5B, b5, h3, nullptr, DM, 0);
  k_final<<<NE, 256, 0, stream>>>(h3, qf, gfin, bfin, Wvec, bvc, out);
}
